// DeltaMemoryHead_29875792511782
// MI455X (gfx1250) — matrix operand order repaired and checked by reading, not compiled, not run on hardware
//
#include <hip/hip_runtime.h>
#include <hip/hip_bf16.h>
#include <math.h>

// ---------------- types ----------------
typedef __bf16 bf16_t;
typedef __attribute__((ext_vector_type(16))) __bf16 v16bf;
typedef __attribute__((ext_vector_type(8)))  __bf16 v8bf;
typedef __attribute__((ext_vector_type(4)))  __bf16 v4bf;
typedef __attribute__((ext_vector_type(8)))  float  v8f;
typedef __attribute__((ext_vector_type(4)))  float  v4f;
typedef __attribute__((ext_vector_type(4)))  unsigned int u32x4;
typedef __attribute__((ext_vector_type(8)))  int i32x8;
typedef __attribute__((ext_vector_type(4)))  int i32x4;

// ---------------- problem constants ----------------
#define B_ROWS 32768
#define D_IN   2048
#define NCAT   576          // 128 Q | 128 K | 128 V | 128 prec | beta | tau | pad -> 36 n-tiles
#define NT1    36           // n-tiles (16 wide) in gemm1
#define KT1    64           // k-tiles (32 deep) in gemm1 (2048/32)

#define WCAT_ELEMS (KT1 * NT1 * 512)   // 64*36*32lanes*16elems = 1,179,648 bf16
#define SFRAG_ELEMS (4 * 8 * 512)      // S: 4 k-tiles * 8 n-tiles       = 16,384 bf16
#define SFRAG_BYTES (SFRAG_ELEMS * 2)  // 32,768 B
#define KSTEP_BYTES (NT1 * 512 * 2)    // one k-step of wfrag = 36,864 B (contiguous)

// reduction buffer layout (floats): [0,128) k_sum | [128,256) prec_sum |
// [256,384) errU_sum | 384 beta_sum | 385 tau_sum
#define RED_N 512

// ---------------- workspace offsets (bytes, all 256B aligned) ----------------
#define OFF_WFRAG 0ull
#define OFF_SFRAG (OFF_WFRAG + (size_t)WCAT_ELEMS * 2)            // 2,359,296
#define OFF_BCAT  (OFF_SFRAG + (size_t)SFRAG_ELEMS * 2)           // +32,768
#define OFF_RED   (OFF_BCAT + (size_t)NCAT * 4)                   // +2,304
#define OFF_Y     (OFF_RED + (size_t)RED_N * 4)                   // +2,048
#define OFF_QK    (OFF_Y + (size_t)B_ROWS * NCAT * 4)             // +75.5 MB
#define OFF_V     (OFF_QK + (size_t)(2 * B_ROWS) * 128 * 2)       // +16.8 MB
#define OFF_R     (OFF_V + (size_t)B_ROWS * 128 * 4)              // +16.8 MB

struct WPtrs {
  const float *Wq, *bq, *Wk, *bk, *Wv, *bv, *Wp, *bp, *Wbe, *bbe, *Wt, *bt, *S;
};

// ---------------- helpers ----------------
__device__ __forceinline__ float wave_sum(float v) {
#pragma unroll
  for (int m = 16; m >= 1; m >>= 1) v += __shfl_xor(v, m, 32);
  return v;
}
__device__ __forceinline__ float sigmoidf_(float x) { return 1.f / (1.f + expf(-x)); }
__device__ __forceinline__ float softplusf_(float x) { return x > 20.f ? x : log1pf(expf(x)); }

// ---------------- TDM: 1-D async copy global -> LDS (Tensor Data Mover) ----------------
// D# per CDNA5 ISA. data_size = 8 bytes;
// tensor_dim0 = tile_dim0 = nbytes/8 (1 row). lds_addr = flat-shared addr low 32 bits
// (flat LDS aperture: LDS address is the low 32 address bits, ISA §10.2).
#if __has_builtin(__builtin_amdgcn_tensor_load_to_lds)
#define HAVE_TDM 1
__device__ __forceinline__ void tdm_load_1d(unsigned int lds_off,
                                            unsigned long long gaddr,
                                            unsigned int nbytes) {
  const unsigned int n8 = nbytes >> 3;  // 8-byte units
  u32x4 g0;
  g0[0] = 1u;                                               // count=1, user mode
  g0[1] = lds_off;                                          // lds_addr
  g0[2] = (unsigned int)(gaddr & 0xffffffffull);            // global_addr lo
  g0[3] = (unsigned int)((gaddr >> 32) & 0x01ffffffull)     // global_addr hi (bits 56:32)
        | (2u << 30);                                       // type = 2 ("image")
  i32x8 g1;
  g1[0] = (int)(3u << 16);                                  // data_size = 8B
  g1[1] = (int)((n8 & 0xffffu) << 16);                      // tensor_dim0[15:0]
  g1[2] = (int)(((n8 >> 16) & 0xffffu) | (1u << 16));       // tensor_dim0[31:16] | tensor_dim1=1
  g1[3] = (int)((n8 & 0xffffu) << 16);                      // tensor_dim1 hi=0 | tile_dim0=n8
  g1[4] = 1;                                                // tile_dim1=1, tile_dim2=0
  g1[5] = (int)n8;                                          // tensor_dim0_stride lo
  g1[6] = 0;                                                // stride hi | dim1_stride lo
  g1[7] = 0;
  i32x4 z = {0, 0, 0, 0};
#if defined(__clang_major__) && __clang_major__ >= 23
  i32x8 z8 = {};
  __builtin_amdgcn_tensor_load_to_lds(g0, g1, z, z, z8, 0);
#else
  __builtin_amdgcn_tensor_load_to_lds(g0, g1, z, z, 0);
#endif
}
#else
#define HAVE_TDM 0
#endif

// ---------------- K0: prep (bf16 fragment-ordered weights, S frags, biases, zero reds) ----
// B-fragment packing used here (contiguous K, matching the A loads; 32x16 K x N): lane 0-15 -> N=lane, K=e (0..15);
// lane 16-31 -> N=lane-16, K=16+e. Each fragment is 32 lanes * 16 bf16 = 512 elems.
__global__ __launch_bounds__(256) void prep_kernel(WPtrs p, bf16_t* __restrict__ wfrag,
                                                   bf16_t* __restrict__ sfrag,
                                                   float* __restrict__ bcat,
                                                   float* __restrict__ red) {
  const int idx = blockIdx.x * 256 + threadIdx.x;
  if (idx < WCAT_ELEMS) {
    const int e = idx & 15, lane = (idx >> 4) & 31, tile = idx >> 9;
    const int nt = tile % NT1, kt = tile / NT1;
    const int n = nt * 16 + (lane & 15);
    const int k = kt * 32 + (lane >> 4) * 16 + e;
    float v = 0.f;
    if (n < 128)       v = p.Wq[k * 128 + n];
    else if (n < 256)  v = p.Wk[k * 128 + (n - 128)];
    else if (n < 384)  v = p.Wv[k * 128 + (n - 256)];
    else if (n < 512)  v = p.Wp[k * 128 + (n - 384)];
    else if (n == 512) v = p.Wbe[k];
    else if (n == 513) v = p.Wt[k];
    wfrag[idx] = (bf16_t)v;
  } else if (idx < WCAT_ELEMS + SFRAG_ELEMS) {
    const int j = idx - WCAT_ELEMS;
    const int e = j & 15, lane = (j >> 4) & 31, tile = j >> 9;
    const int nt = tile & 7, kt = tile >> 3;
    const int n = nt * 16 + (lane & 15);
    const int k = kt * 32 + (lane >> 4) * 16 + e;
    sfrag[j] = (bf16_t)p.S[k * 128 + n];
  } else if (idx < WCAT_ELEMS + SFRAG_ELEMS + NCAT) {
    const int c = idx - (WCAT_ELEMS + SFRAG_ELEMS);
    float b = 0.f;
    if (c < 128)       b = p.bq[c];
    else if (c < 256)  b = p.bk[c - 128];
    else if (c < 384)  b = p.bv[c - 256];
    else if (c < 512)  b = p.bp[c - 384];
    else if (c == 512) b = p.bbe[0];
    else if (c == 513) b = p.bt[0];
    bcat[c] = b;
  } else if (idx < WCAT_ELEMS + SFRAG_ELEMS + NCAT + RED_N) {
    red[idx - (WCAT_ELEMS + SFRAG_ELEMS + NCAT)] = 0.f;
  }
}

// ---------------- K1: fused projection GEMM  Y = bf16(x) @ Wcat + bcat ----------------
// Block: 512 threads = 16 waves = 4 M-subtiles (16 rows each) x 4 N-groups (9 n-tiles each).
// B fragments double-buffered in LDS via TDM (one DMA per block per k-step, shared by all
// 16 waves) and double-buffered again in registers across the 9-tile WMMA chain.
// A fragments: global f32 x prefetched one k-step ahead, converted to bf16 in the shadow
// of the in-flight loads.
__global__ __launch_bounds__(512) void gemm1_kernel(const float* __restrict__ x,
                                                    const bf16_t* __restrict__ wfrag,
                                                    const float* __restrict__ bcat,
                                                    float* __restrict__ Y) {
  __shared__ __align__(16) unsigned char smem[2][KSTEP_BYTES];  // 72 KB double buffer
  const int tid = threadIdx.x;
  const int lane = tid & 31, w = tid >> 5;
  const int mw = w & 3, ng = w >> 2;
  const int rowbase = blockIdx.x * 64 + mw * 16;
  const int arow = rowbase + (lane & 15);
  const int koff = (lane >> 4) * 16;
  const float* xr = x + (size_t)arow * D_IN + koff;
  const unsigned long long wbase = (unsigned long long)(uintptr_t)wfrag;

#if HAVE_TDM
  const unsigned int lds0 = (unsigned int)(uintptr_t)&smem[0][0];
  const unsigned int lds1 = (unsigned int)(uintptr_t)&smem[1][0];
  if (w == 0) {
    tdm_load_1d(lds0, wbase, KSTEP_BYTES);
    __builtin_amdgcn_s_wait_tensorcnt(0);
  }
  __syncthreads();
#else
  for (int i = tid; i < KSTEP_BYTES / 16; i += 512)
    ((u32x4*)&smem[0][0])[i] = ((const u32x4*)wfrag)[i];
  __syncthreads();
#endif

  // A prefetch registers (k-step 0)
  v4f a0 = *(const v4f*)(xr);
  v4f a1 = *(const v4f*)(xr + 4);
  v4f a2 = *(const v4f*)(xr + 8);
  v4f a3 = *(const v4f*)(xr + 12);

  v8f acc[9] = {};
  for (int kt = 0; kt < KT1; ++kt) {
    const int cur = kt & 1;
    // kick off DMA for next k-step into the other buffer
    if (kt + 1 < KT1) {
#if HAVE_TDM
      if (w == 0)
        tdm_load_1d(cur ? lds0 : lds1,
                    wbase + (size_t)(kt + 1) * KSTEP_BYTES, KSTEP_BYTES);
#else
      const u32x4* src = (const u32x4*)(wfrag + (size_t)(kt + 1) * (KSTEP_BYTES / 2));
      u32x4* dst = (u32x4*)&smem[cur ^ 1][0];
      for (int i = tid; i < KSTEP_BYTES / 16; i += 512) dst[i] = src[i];
#endif
    }
    // Convert the already-arrived A data (K order used here, same as wfrag: lane<16 row=lane,
    // K=0..15; lane>=16 row=lane-16, K=16..31)
    v16bf a;
#pragma unroll
    for (int i = 0; i < 4; ++i) {
      a[i]      = (bf16_t)a0[i];
      a[4 + i]  = (bf16_t)a1[i];
      a[8 + i]  = (bf16_t)a2[i];
      a[12 + i] = (bf16_t)a3[i];
    }
    // Prefetch next k-step's A while this k-step's WMMAs run
    if (kt + 1 < KT1) {
      const float* xp = xr + (kt + 1) * 32;
      a0 = *(const v4f*)(xp);
      a1 = *(const v4f*)(xp + 4);
      a2 = *(const v4f*)(xp + 8);
      a3 = *(const v4f*)(xp + 12);
    }
    // B fragments from LDS, register double-buffered: ds_load for tile t+1 is issued
    // before the WMMA for tile t, hiding LDS latency behind the matrix op.
    const unsigned char* bufp = &smem[cur][0] + (size_t)(ng * 9) * 1024 + (size_t)lane * 32;
    v16bf bcur = *(const v16bf*)(bufp);
#pragma unroll
    for (int t = 0; t < 9; ++t) {
      v16bf bnext;
      if (t < 8) bnext = *(const v16bf*)(bufp + (t + 1) * 1024);
      acc[t] = __builtin_amdgcn_wmma_f32_16x16x32_bf16(false, a, false, bcur,
                                                       (short)0, acc[t], false, false);
      bcur = bnext;
    }
#if HAVE_TDM
    if (w == 0 && kt + 1 < KT1) __builtin_amdgcn_s_wait_tensorcnt(0);
#endif
    __syncthreads();
  }
  // C/D layout: vgpr j: lanes 0-15 -> M=j, N=lane; lanes 16-31 -> M=j+8, N=lane-16.
  const int ncol0 = (ng * 9) * 16 + (lane & 15);
  const int mofs = (lane < 16) ? 0 : 8;
#pragma unroll
  for (int t = 0; t < 9; ++t) {
    const int n = ncol0 + t * 16;
    const float bias = bcat[n];
#pragma unroll
    for (int j = 0; j < 8; ++j)
      Y[(size_t)(rowbase + mofs + j) * NCAT + n] = acc[t][j] + bias;
  }
}

// ---------------- K2: per-row normalize/activations, emit bf16 q|k, accumulate means ----
__global__ __launch_bounds__(256) void rowstage_kernel(const float* __restrict__ Y,
                                                       bf16_t* __restrict__ qk,
                                                       float* __restrict__ vbuf,
                                                       float* __restrict__ out_prec,
                                                       float* __restrict__ red) {
  __shared__ float s_ksum[128], s_psum[128], s_bt[2];
  const int tid = threadIdx.x;
  if (tid < 128) { s_ksum[tid] = 0.f; s_psum[tid] = 0.f; }
  if (tid < 2) s_bt[tid] = 0.f;
  __syncthreads();
  const int lane = tid & 31, w = tid >> 5;
  const int r = blockIdx.x * 8 + w;
  const float* y = Y + (size_t)r * NCAT;
  const int c = lane * 4;
  // q = normalize(x@Wq + bq)
  v4f q = *(const v4f*)(y + c);
  float ss = wave_sum(q[0] * q[0] + q[1] * q[1] + q[2] * q[2] + q[3] * q[3]);
  float inv = 1.f / fmaxf(sqrtf(ss), 1e-12f);
  v4bf qb;
#pragma unroll
  for (int i = 0; i < 4; ++i) qb[i] = (bf16_t)(q[i] * inv);
  *(v4bf*)(qk + (size_t)r * 128 + c) = qb;
  // k = normalize(x@Wk + bk)
  v4f kk = *(const v4f*)(y + 128 + c);
  ss = wave_sum(kk[0] * kk[0] + kk[1] * kk[1] + kk[2] * kk[2] + kk[3] * kk[3]);
  inv = 1.f / fmaxf(sqrtf(ss), 1e-12f);
  v4bf kb; float kn[4];
#pragma unroll
  for (int i = 0; i < 4; ++i) { kn[i] = kk[i] * inv; kb[i] = (bf16_t)kn[i]; }
  *(v4bf*)(qk + (size_t)(B_ROWS + r) * 128 + c) = kb;
#pragma unroll
  for (int i = 0; i < 4; ++i) atomicAdd(&s_ksum[c + i], kn[i]);
  // v = x@Wv + bv
  v4f vv = *(const v4f*)(y + 256 + c);
  *(v4f*)(vbuf + (size_t)r * 128 + c) = vv;
  // prec = softplus(x@Wprec + bprec) + 0.01
  v4f pp = *(const v4f*)(y + 384 + c);
  v4f pr;
#pragma unroll
  for (int i = 0; i < 4; ++i) pr[i] = softplusf_(pp[i]) + 0.01f;
  *(v4f*)(out_prec + (size_t)r * 128 + c) = pr;
#pragma unroll
  for (int i = 0; i < 4; ++i) atomicAdd(&s_psum[c + i], pr[i]);
  if (lane == 0) {
    atomicAdd(&s_bt[0], sigmoidf_(y[512]));
    atomicAdd(&s_bt[1], 1.f + 19.f * sigmoidf_(y[513]));
  }
  __syncthreads();
  if (tid < 128) { atomicAdd(&red[tid], s_ksum[tid]); atomicAdd(&red[128 + tid], s_psum[tid]); }
  if (tid == 0) { atomicAdd(&red[384], s_bt[0]); atomicAdd(&red[385], s_bt[1]); }
}

// ---------------- K3: R = [q;k] @ S  (2B x 128) @ (128 x 128), bf16 WMMA ----------------
// S fragments (32 KB) staged once per block into LDS via TDM; A fragments hoisted.
__global__ __launch_bounds__(256) void gemm2_kernel(const bf16_t* __restrict__ qk,
                                                    const bf16_t* __restrict__ sfrag,
                                                    float* __restrict__ R) {
  __shared__ __align__(16) unsigned char ssm[SFRAG_BYTES];  // 32 KB
  const int tid = threadIdx.x, lane = tid & 31, w = tid >> 5;
#if HAVE_TDM
  if (w == 0) {
    tdm_load_1d((unsigned int)(uintptr_t)&ssm[0],
                (unsigned long long)(uintptr_t)sfrag, SFRAG_BYTES);
    __builtin_amdgcn_s_wait_tensorcnt(0);
  }
  __syncthreads();
#else
  for (int i = tid; i < SFRAG_BYTES / 16; i += 256)
    ((u32x4*)&ssm[0])[i] = ((const u32x4*)sfrag)[i];
  __syncthreads();
#endif
  const int rowbase = blockIdx.x * 128 + w * 16;
  const int arow = rowbase + (lane & 15);
  const int koff = (lane >> 4) * 16;
  const bf16_t* ar = qk + (size_t)arow * 128 + koff;
  // hoist all 4 A fragments (K = 128 = 4 k-steps)
  v16bf afr[4];
#pragma unroll
  for (int kt = 0; kt < 4; ++kt) {
    v8bf g0 = *(const v8bf*)(ar + kt * 32);
    v8bf g1 = *(const v8bf*)(ar + kt * 32 + 8);
#pragma unroll
    for (int i = 0; i < 8; ++i) { afr[kt][i] = g0[i]; afr[kt][8 + i] = g1[i]; }
  }
  v8f acc[8] = {};
#pragma unroll
  for (int kt = 0; kt < 4; ++kt) {
    const unsigned char* bufp = &ssm[0] + (size_t)kt * 8 * 1024 + (size_t)lane * 32;
    v16bf bcur = *(const v16bf*)(bufp);
#pragma unroll
    for (int t = 0; t < 8; ++t) {
      v16bf bnext;
      if (t < 7) bnext = *(const v16bf*)(bufp + (t + 1) * 1024);
      acc[t] = __builtin_amdgcn_wmma_f32_16x16x32_bf16(false, afr[kt], false, bcur,
                                                       (short)0, acc[t], false, false);
      bcur = bnext;
    }
  }
  const int mofs = (lane < 16) ? 0 : 8;
#pragma unroll
  for (int t = 0; t < 8; ++t) {
    const int n = t * 16 + (lane & 15);
#pragma unroll
    for (int j = 0; j < 8; ++j)
      R[(size_t)(rowbase + mofs + j) * 128 + n] = acc[t][j];
  }
}

// ---------------- K4: error, F_mean, clipped-error column means ----------------
__global__ __launch_bounds__(256) void finstage_kernel(const float* __restrict__ target,
                                                       const float* __restrict__ R,
                                                       const float* __restrict__ vbuf,
                                                       const float* __restrict__ out_prec,
                                                       float* __restrict__ out_err,
                                                       float* __restrict__ out_F,
                                                       float* __restrict__ red) {
  __shared__ float s_esum[128];
  const int tid = threadIdx.x;
  if (tid < 128) s_esum[tid] = 0.f;
  __syncthreads();
  const int lane = tid & 31, w = tid >> 5;
  const int r = blockIdx.x * 8 + w;
  const int c = lane * 4;
  v4f tg = *(const v4f*)(target + (size_t)r * 128 + c);
  v4f rt = *(const v4f*)(R + (size_t)r * 128 + c);
  v4f pd = *(const v4f*)(R + (size_t)(B_ROWS + r) * 128 + c);
  v4f vv = *(const v4f*)(vbuf + (size_t)r * 128 + c);
  v4f pr = *(const v4f*)(out_prec + (size_t)r * 128 + c);
  v4f er; float f = 0.f;
#pragma unroll
  for (int i = 0; i < 4; ++i) {
    er[i] = tg[i] - rt[i];
    f += pr[i] * er[i] * er[i] - logf(pr[i]);
  }
  *(v4f*)(out_err + (size_t)r * 128 + c) = er;
  f = wave_sum(f);
  if (lane == 0) out_F[r] = f * (1.f / 128.f);
#pragma unroll
  for (int i = 0; i < 4; ++i) {
    float eu = 2.f * tanhf((pd[i] - vv[i]) * 0.5f);  // max_err * tanh(d / max_err)
    atomicAdd(&s_esum[c + i], eu);
  }
  __syncthreads();
  if (tid < 128) atomicAdd(&red[256 + tid], s_esum[tid]);
}

// ---------------- K5: S_new / momentum_new ----------------
__global__ __launch_bounds__(256) void update_kernel(const float* __restrict__ S,
                                                     const float* __restrict__ momentum,
                                                     const float* __restrict__ red,
                                                     float* __restrict__ out_Snew,
                                                     float* __restrict__ out_mom) {
  const int idx = blockIdx.x * 256 + threadIdx.x;  // 0..16383
  const int i = idx >> 7, j = idx & 127;
  const float invB = 1.f / (float)B_ROWS;
  const float kmean = red[i] * invB;
  const float pmean = red[128 + j] * invB;
  const float emean = red[256 + j] * invB;
  const float bmean = red[384] * invB;
  const float tmean = red[385] * invB;
  const float decay = 1.f / (1.f + expf(0.1f)) + 0.5f;  // sigmoid(-0.1) + 0.5
  const float eff = 1.f - (1.f - decay) / tmean;
  const float upd = kmean * (bmean * pmean * emean);
  const float mnew = 0.9f * momentum[idx] + 0.1f * upd;
  out_mom[idx] = mnew;
  out_Snew[idx] = eff * S[idx] - 0.01f * mnew;
}

// ---------------- launch ----------------
extern "C" void kernel_launch(void* const* d_in, const int* in_sizes, int n_in,
                              void* d_out, int out_size, void* d_ws, size_t ws_size,
                              hipStream_t stream) {
  const float* x      = (const float*)d_in[0];
  const float* target = (const float*)d_in[1];
  WPtrs p;
  p.Wk  = (const float*)d_in[2];  p.bk  = (const float*)d_in[3];
  p.Wv  = (const float*)d_in[4];  p.bv  = (const float*)d_in[5];
  p.Wq  = (const float*)d_in[6];  p.bq  = (const float*)d_in[7];
  p.Wbe = (const float*)d_in[8];  p.bbe = (const float*)d_in[9];
  p.Wp  = (const float*)d_in[10]; p.bp  = (const float*)d_in[11];
  p.Wt  = (const float*)d_in[12]; p.bt  = (const float*)d_in[13];
  p.S   = (const float*)d_in[14];
  const float* momentum = (const float*)d_in[15];

  char* ws = (char*)d_ws;
  bf16_t* wfrag = (bf16_t*)(ws + OFF_WFRAG);
  bf16_t* sfrag = (bf16_t*)(ws + OFF_SFRAG);
  float* bcat   = (float*)(ws + OFF_BCAT);
  float* red    = (float*)(ws + OFF_RED);
  float* Y      = (float*)(ws + OFF_Y);
  bf16_t* qk    = (bf16_t*)(ws + OFF_QK);
  float* vbuf   = (float*)(ws + OFF_V);
  float* R      = (float*)(ws + OFF_R);

  float* out_F    = (float*)d_out;
  float* out_prec = out_F + B_ROWS;
  float* out_err  = out_prec + (size_t)B_ROWS * 128;
  float* out_Snew = out_err + (size_t)B_ROWS * 128;
  float* out_mom  = out_Snew + 128 * 128;

  const int prep_total = WCAT_ELEMS + SFRAG_ELEMS + NCAT + RED_N;
  prep_kernel<<<(prep_total + 255) / 256, 256, 0, stream>>>(p, wfrag, sfrag, bcat, red);
  gemm1_kernel<<<B_ROWS / 64, 512, 0, stream>>>(x, wfrag, bcat, Y);
  rowstage_kernel<<<B_ROWS / 8, 256, 0, stream>>>(Y, qk, vbuf, out_prec, red);
  gemm2_kernel<<<(2 * B_ROWS) / 128, 256, 0, stream>>>(qk, sfrag, R);
  finstage_kernel<<<B_ROWS / 8, 256, 0, stream>>>(target, R, vbuf, out_prec, out_err, out_F, red);
  update_kernel<<<(128 * 128) / 256, 256, 0, stream>>>(p.S, momentum, red, out_Snew, out_mom);
}
